// MultiScaleTransformerLayer_59785944761262
// MI455X (gfx1250) — compile-verified, matrix operand order checked by reading, not run on hardware
//
#include <hip/hip_runtime.h>
#include <hip/hip_bf16.h>

typedef _Float16 h16;
typedef __attribute__((ext_vector_type(16))) _Float16 v16h;
typedef __attribute__((ext_vector_type(8)))  float    v8f;
typedef __attribute__((ext_vector_type(4)))  int      i32x4;

constexpr int Bc = 32, Tc = 512, Hc = 1024, NHc = 16, HDc = 64, H4c = 4096;
constexpr int Mrows = Bc * Tc; // 16384

// --------------------------------------------------------------------------
// CDNA5 async global->LDS path (guarded: falls back if unavailable)
// --------------------------------------------------------------------------
#if __has_builtin(__builtin_amdgcn_global_load_async_to_lds_b128) && \
    __has_builtin(__builtin_amdgcn_s_wait_asynccnt)
#define USE_ASYNC_LDS 1
#else
#define USE_ASYNC_LDS 0
#endif

#if USE_ASYNC_LDS
__device__ __forceinline__ void async_cp16B(const h16* gsrc, h16* ldst)
{
    __builtin_amdgcn_global_load_async_to_lds_b128(
        (__attribute__((address_space(1))) i32x4*)gsrc,
        (__attribute__((address_space(3))) i32x4*)ldst, 0, 0);
}
#endif

// ---------------------------------------------------------------------------
// LayerNorm (fp32 in) -> f16 out
// ---------------------------------------------------------------------------
__global__ __launch_bounds__(256)
void ln_cast_kernel(const float* __restrict__ x, const float* __restrict__ g,
                    const float* __restrict__ be, h16* __restrict__ out)
{
    __shared__ float red[256];
    const int row = blockIdx.x;
    const float* xr = x + (size_t)row * Hc;
    float v[4];
    float s = 0.f;
#pragma unroll
    for (int i = 0; i < 4; ++i) { v[i] = xr[threadIdx.x + i * 256]; s += v[i]; }
    red[threadIdx.x] = s; __syncthreads();
    for (int st = 128; st > 0; st >>= 1) {
        if (threadIdx.x < st) red[threadIdx.x] += red[threadIdx.x + st];
        __syncthreads();
    }
    const float mu = red[0] * (1.f / Hc);
    __syncthreads();
    s = 0.f;
#pragma unroll
    for (int i = 0; i < 4; ++i) { float d = v[i] - mu; s += d * d; }
    red[threadIdx.x] = s; __syncthreads();
    for (int st = 128; st > 0; st >>= 1) {
        if (threadIdx.x < st) red[threadIdx.x] += red[threadIdx.x + st];
        __syncthreads();
    }
    const float rstd = rsqrtf(red[0] * (1.f / Hc) + 1e-5f);
    h16* outr = out + (size_t)row * Hc;
#pragma unroll
    for (int i = 0; i < 4; ++i) {
        const int c = threadIdx.x + i * 256;
        outr[c] = (h16)((v[i] - mu) * rstd * g[c] + be[c]);
    }
}

// ---------------------------------------------------------------------------
// Weight transpose + cast: W fp32 [K,N] -> Wt f16 [N,K]
// ---------------------------------------------------------------------------
__global__ __launch_bounds__(256)
void wtrans_kernel(const float* __restrict__ W, h16* __restrict__ Wt, int K, int N)
{
    __shared__ float tile[32][33];
    const int n0 = blockIdx.x * 32, k0 = blockIdx.y * 32;
    const int tx = threadIdx.x, ty = threadIdx.y; // 32 x 8
#pragma unroll
    for (int i = 0; i < 4; ++i)
        tile[ty + i * 8][tx] = W[(size_t)(k0 + ty + i * 8) * N + n0 + tx];
    __syncthreads();
#pragma unroll
    for (int i = 0; i < 4; ++i)
        Wt[(size_t)(n0 + ty + i * 8) * K + k0 + tx] = (h16)tile[tx][ty + i * 8];
}

// ---------------------------------------------------------------------------
// WMMA GEMM: C[M,N] = A[M,K](f16,row) x Bt[N,K](f16 = B col-major) + bias
// 128x128 tile, BK=64, double-buffered LDS, ONE barrier per K-step.
// 256 threads = 8 waves (4 in M x 2 in N); 16 v_wmma per K-step per wave.
// Tile staging uses async global->LDS DMA (ASYNCcnt) when available, else
// register staging; next tile always overlaps this step's WMMAs.
// ---------------------------------------------------------------------------
enum { EPI_NONE = 0, EPI_GELU = 1, EPI_RESID = 2 };

template <int EPI, bool OUT_F32, bool OUT_F16>
__global__ __launch_bounds__(256)
void gemm_f16_kernel(const h16* __restrict__ A, const h16* __restrict__ Bt,
                     const float* __restrict__ bias, const float* __restrict__ resid,
                     float* __restrict__ Cf, h16* __restrict__ Ch,
                     int Msz, int Nsz, int Ksz)
{
    constexpr int LDT = 72; // halves per LDS row (144B: conflict-free, 16B aligned)
    __shared__ __align__(16) h16 As[2][128 * LDT];
    __shared__ __align__(16) h16 Bs[2][128 * LDT];

    const int tid  = threadIdx.x;
    const int lane = tid & 31, wid = tid >> 5;
    const int L = lane & 15, gsel = lane >> 4;
    const int wm = (wid & 3) * 32;  // wave row offset within tile
    const int wn = (wid >> 2) * 64; // wave col offset within tile
    const size_t mBase = (size_t)blockIdx.y * 128;
    const size_t nBase = (size_t)blockIdx.x * 128;

    // per-thread staging coords: 4 chunks of 8 halves per tile (1024 chunks total)
    int srow[4], sc8[4];
#pragma unroll
    for (int i = 0; i < 4; ++i) {
        const int cid = i * 256 + tid;
        srow[i] = cid >> 3;
        sc8[i]  = (cid & 7) << 3;
    }
    const h16* aSrc = A  + (mBase) * (size_t)Ksz;
    const h16* bSrc = Bt + (nBase) * (size_t)Ksz;

    v8f acc[2][4];
    const v8f vz = {};
#pragma unroll
    for (int i = 0; i < 2; ++i)
#pragma unroll
        for (int j = 0; j < 4; ++j) acc[i][j] = vz;

    const int kIters = Ksz >> 6;

#if USE_ASYNC_LDS
    // prologue: async DMA tile 0 -> LDS buf 0
#pragma unroll
    for (int i = 0; i < 4; ++i) {
        async_cp16B(aSrc + (size_t)srow[i] * Ksz + sc8[i], As[0] + srow[i] * LDT + sc8[i]);
        async_cp16B(bSrc + (size_t)srow[i] * Ksz + sc8[i], Bs[0] + srow[i] * LDT + sc8[i]);
    }
    __builtin_amdgcn_s_wait_asynccnt(0);
#else
    // prologue: tile 0 -> regs -> LDS buf 0
    float4 ra[4], rb[4];
#pragma unroll
    for (int i = 0; i < 4; ++i) {
        ra[i] = *(const float4*)(aSrc + (size_t)srow[i] * Ksz + sc8[i]);
        rb[i] = *(const float4*)(bSrc + (size_t)srow[i] * Ksz + sc8[i]);
    }
#pragma unroll
    for (int i = 0; i < 4; ++i) {
        *(float4*)(As[0] + srow[i] * LDT + sc8[i]) = ra[i];
        *(float4*)(Bs[0] + srow[i] * LDT + sc8[i]) = rb[i];
    }
#endif

    for (int kb = 0; kb < kIters; ++kb) {
        const int cur = kb & 1, nxt = cur ^ 1;
        // pull the K+2 tile toward the caches (global_prefetch_b8)
        if (kb + 2 < kIters) {
            const int k0 = (kb + 2) << 6;
            __builtin_prefetch(aSrc + (size_t)srow[0] * Ksz + k0 + sc8[0], 0, 3);
            __builtin_prefetch(bSrc + (size_t)srow[0] * Ksz + k0 + sc8[0], 0, 3);
        }
#if !USE_ASYNC_LDS
        // issue next tile's global loads (retire behind this step's WMMAs)
        float4 na[4], nb[4];
        if (kb + 1 < kIters) {
            const int k0 = (kb + 1) << 6;
#pragma unroll
            for (int i = 0; i < 4; ++i) {
                na[i] = *(const float4*)(aSrc + (size_t)srow[i] * Ksz + k0 + sc8[i]);
                nb[i] = *(const float4*)(bSrc + (size_t)srow[i] * Ksz + k0 + sc8[i]);
            }
        }
#endif
        __syncthreads(); // buf[cur] published; prior reads of buf[nxt] retired

#if USE_ASYNC_LDS
        // async DMA next tile straight into buf[nxt]; overlaps WMMAs below
        if (kb + 1 < kIters) {
            const int k0 = (kb + 1) << 6;
#pragma unroll
            for (int i = 0; i < 4; ++i) {
                async_cp16B(aSrc + (size_t)srow[i] * Ksz + k0 + sc8[i],
                            As[nxt] + srow[i] * LDT + sc8[i]);
                async_cp16B(bSrc + (size_t)srow[i] * Ksz + k0 + sc8[i],
                            Bs[nxt] + srow[i] * LDT + sc8[i]);
            }
        }
#endif

        // compute on buf[cur]: 2 K-chunks of 32 -> 16 WMMAs
#pragma unroll
        for (int kc = 0; kc < 2; ++kc) {
            v16h af[2], bf[4];
#pragma unroll
            for (int mt = 0; mt < 2; ++mt) {
                const h16* p = As[cur] + (wm + mt * 16 + L) * LDT + kc * 32;
                float4* d = (float4*)&af[mt];
                d[0] = *(const float4*)(p + 8 * gsel);
                d[1] = *(const float4*)(p + 16 + 8 * gsel);
            }
#pragma unroll
            for (int nt = 0; nt < 4; ++nt) {
                const h16* p = Bs[cur] + (wn + nt * 16 + L) * LDT + kc * 32;
                float4* d = (float4*)&bf[nt];
                d[0] = *(const float4*)(p + 8 * gsel);
                d[1] = *(const float4*)(p + 16 + 8 * gsel);
            }
#pragma unroll
            for (int mt = 0; mt < 2; ++mt)
#pragma unroll
                for (int nt = 0; nt < 4; ++nt)
                    acc[mt][nt] = __builtin_amdgcn_wmma_f32_16x16x32_f16(
                        false, af[mt], false, bf[nt], (short)0, acc[mt][nt], false, false);
        }

#if USE_ASYNC_LDS
        if (kb + 1 < kIters)
            __builtin_amdgcn_s_wait_asynccnt(0); // my DMA into buf[nxt] landed
#else
        if (kb + 1 < kIters) {
#pragma unroll
            for (int i = 0; i < 4; ++i) {
                *(float4*)(As[nxt] + srow[i] * LDT + sc8[i]) = na[i];
                *(float4*)(Bs[nxt] + srow[i] * LDT + sc8[i]) = nb[i];
            }
        }
#endif
    }

    // Epilogue. C layout: row = r + 8*gsel, col = L (per 16x16 tile).
#pragma unroll
    for (int mt = 0; mt < 2; ++mt) {
#pragma unroll
        for (int nt = 0; nt < 4; ++nt) {
            const size_t col = nBase + wn + nt * 16 + L;
            const float bvv = bias[col];
#pragma unroll
            for (int r = 0; r < 8; ++r) {
                const size_t row = mBase + wm + mt * 16 + r + 8 * gsel;
                float v = acc[mt][nt][r] + bvv;
                if constexpr (EPI == EPI_GELU)
                    v = 0.5f * v * (1.f + erff(v * 0.70710678118654752f));
                if constexpr (EPI == EPI_RESID)
                    v += resid[row * (size_t)Nsz + col];
                if constexpr (OUT_F32) Cf[row * (size_t)Nsz + col] = v;
                if constexpr (OUT_F16) Ch[row * (size_t)Nsz + col] = (h16)v;
            }
        }
    }
}

// ---------------------------------------------------------------------------
// RoPE + repack: q,k,v fp32 [B*T, H] -> qh,kh f16 [B*NH, T, 64], vT f16 [B*NH, 64, T]
// ---------------------------------------------------------------------------
__global__ __launch_bounds__(256)
void rope_pack_kernel(const float* __restrict__ q, const float* __restrict__ k,
                      const float* __restrict__ v, h16* __restrict__ qh,
                      h16* __restrict__ kh, h16* __restrict__ vT)
{
    const int bt = blockIdx.x;
    const int b = bt >> 9, t = bt & (Tc - 1);
    const float* qr = q + (size_t)bt * Hc;
    const float* kr = k + (size_t)bt * Hc;
    const float* vr = v + (size_t)bt * Hc;
#pragma unroll
    for (int i = 0; i < 4; ++i) {
        const int e = threadIdx.x + i * 256;
        const int h = e >> 6, d = e & 63;
        const int j = d & 31;
        // 10000^(-j/32) = exp(-j * ln(10000)/32)
        const float ang = (float)t * __expf(-(float)j * 0.28782313662425572f);
        float sn, cs;
        __sincosf(ang, &sn, &cs);
        const int base = h * 64;
        const float rq = (d < 32) ? -qr[base + d + 32] : qr[base + d - 32];
        const float rk = (d < 32) ? -kr[base + d + 32] : kr[base + d - 32];
        const float qv = qr[e] * cs + rq * sn;
        const float kv = kr[e] * cs + rk * sn;
        const size_t bh = (size_t)b * NHc + h;
        qh[(bh * Tc + t) * HDc + d] = (h16)qv;
        kh[(bh * Tc + t) * HDc + d] = (h16)kv;
        vT[(bh * HDc + d) * Tc + t] = (h16)vr[e];
    }
}

// ---------------------------------------------------------------------------
// Flash attention: one block per (b*NH, 64-query-block). 128 threads = 4 waves,
// each wave owns 16 query rows. K-blocks of 64 keys; online softmax.
// ---------------------------------------------------------------------------
__global__ __launch_bounds__(128)
void attn_kernel(const h16* __restrict__ qh, const h16* __restrict__ kh,
                 const h16* __restrict__ vT, h16* __restrict__ attn)
{
    constexpr int LDK = 72; // 144B rows: conflict-free 16B/lane reads
    __shared__ __align__(16) h16 ks[64 * LDK];
    __shared__ __align__(16) h16 vs[64 * LDK];
    __shared__ __align__(16) h16 ps[4 * 16 * LDK];

    const int bh = blockIdx.x;
    const int b = bh / NHc, h = bh % NHc;
    const int qBase = blockIdx.y * 64;
    const int tid = threadIdx.x;
    const int lane = tid & 31, w = tid >> 5;
    const int L = lane & 15, gsel = lane >> 4;

    // Q fragments for this wave's 16 rows (A-layout: lane holds row L)
    v16h aq0, aq1;
    {
        const size_t qrow = (size_t)bh * Tc + qBase + w * 16 + L;
        const float4* qp = (const float4*)(qh + qrow * HDc);
        float4* d0 = (float4*)&aq0; float4* d1 = (float4*)&aq1;
        d0[0] = qp[gsel];     d0[1] = qp[2 + gsel];
        d1[0] = qp[4 + gsel]; d1[1] = qp[6 + gsel];
    }

    float m[8], lsum[8];
    v8f o[4];
    const v8f vz = {};
#pragma unroll
    for (int r = 0; r < 8; ++r) { m[r] = -1e30f; lsum[r] = 0.f; }
#pragma unroll
    for (int nt = 0; nt < 4; ++nt) o[nt] = vz;

    const float scale = 0.125f; // 1/sqrt(64)

    for (int kb = 0; kb < Tc / 64; ++kb) {
        const int kBase = kb * 64;
        __syncthreads();
        // stage K tile [64 keys x 64 dims] and V^T tile [64 dims x 64 keys]
#pragma unroll
        for (int i = 0; i < 4; ++i) {
            const int cid = i * 128 + tid;
            const int row = cid >> 3, c8 = (cid & 7) << 3;
            *(float4*)(ks + row * LDK + c8) =
                *(const float4*)(kh + ((size_t)bh * Tc + kBase + row) * HDc + c8);
            *(float4*)(vs + row * LDK + c8) =
                *(const float4*)(vT + ((size_t)bh * HDc + row) * Tc + kBase + c8);
        }
        __syncthreads();

        // S = q @ k^T : per wave 16x64, via 4 n-tiles x 2 WMMAs over HD=64
        v8f sacc[4];
#pragma unroll
        for (int nt = 0; nt < 4; ++nt) {
            const h16* p = ks + (nt * 16 + L) * LDK;
            v16h blo, bhi;
            float4* dl = (float4*)&blo; float4* dh = (float4*)&bhi;
            dl[0] = *(const float4*)(p + 8 * gsel);
            dl[1] = *(const float4*)(p + 16 + 8 * gsel);
            dh[0] = *(const float4*)(p + 32 + 8 * gsel);
            dh[1] = *(const float4*)(p + 48 + 8 * gsel);
            v8f c = vz;
            c = __builtin_amdgcn_wmma_f32_16x16x32_f16(false, aq0, false, blo, (short)0, c, false, false);
            c = __builtin_amdgcn_wmma_f32_16x16x32_f16(false, aq1, false, bhi, (short)0, c, false, false);
            sacc[nt] = c;
        }

        // online softmax: rows r+8*gsel, cols across 16 lanes of same half
#pragma unroll
        for (int r = 0; r < 8; ++r) {
            float mx = -1e30f;
#pragma unroll
            for (int nt = 0; nt < 4; ++nt) mx = fmaxf(mx, sacc[nt][r] * scale);
#pragma unroll
            for (int off = 1; off < 16; off <<= 1) mx = fmaxf(mx, __shfl_xor(mx, off, 32));
            const float mnew = fmaxf(m[r], mx);
            const float alpha = __expf(m[r] - mnew);
            m[r] = mnew;
            float rs = 0.f;
#pragma unroll
            for (int nt = 0; nt < 4; ++nt) {
                const float pv = __expf(sacc[nt][r] * scale - mnew);
                sacc[nt][r] = pv;
                rs += pv;
            }
#pragma unroll
            for (int off = 1; off < 16; off <<= 1) rs += __shfl_xor(rs, off, 32);
            lsum[r] = lsum[r] * alpha + rs;
#pragma unroll
            for (int nt = 0; nt < 4; ++nt) o[nt][r] *= alpha;
        }

        // stage P (C-layout) into LDS, reload as A-fragments
        h16* pw = ps + w * 16 * LDK;
#pragma unroll
        for (int nt = 0; nt < 4; ++nt)
#pragma unroll
            for (int r = 0; r < 8; ++r)
                pw[(r + 8 * gsel) * LDK + nt * 16 + L] = (h16)sacc[nt][r];
        __syncthreads();
        v16h ap0, ap1;
        {
            const h16* p = pw + L * LDK;
            float4* d0 = (float4*)&ap0; float4* d1 = (float4*)&ap1;
            d0[0] = *(const float4*)(p + 8 * gsel);
            d0[1] = *(const float4*)(p + 16 + 8 * gsel);
            d1[0] = *(const float4*)(p + 32 + 8 * gsel);
            d1[1] = *(const float4*)(p + 48 + 8 * gsel);
        }
        // O += P @ V (B-frags from V^T rows = head dims)
#pragma unroll
        for (int nt = 0; nt < 4; ++nt) {
            const h16* p = vs + (nt * 16 + L) * LDK;
            v16h blo, bhi;
            float4* dl = (float4*)&blo; float4* dh = (float4*)&bhi;
            dl[0] = *(const float4*)(p + 8 * gsel);
            dl[1] = *(const float4*)(p + 16 + 8 * gsel);
            dh[0] = *(const float4*)(p + 32 + 8 * gsel);
            dh[1] = *(const float4*)(p + 48 + 8 * gsel);
            o[nt] = __builtin_amdgcn_wmma_f32_16x16x32_f16(false, ap0, false, blo, (short)0, o[nt], false, false);
            o[nt] = __builtin_amdgcn_wmma_f32_16x16x32_f16(false, ap1, false, bhi, (short)0, o[nt], false, false);
        }
    }

    // normalize, write head-interleaved f16 [B, T, H]
#pragma unroll
    for (int r = 0; r < 8; ++r) {
        const float inv = 1.f / lsum[r];
        const int qrow = qBase + w * 16 + r + 8 * gsel;
        const size_t rowBase = ((size_t)b * Tc + qrow) * Hc + h * HDc;
#pragma unroll
        for (int nt = 0; nt < 4; ++nt)
            attn[rowBase + nt * 16 + L] = (h16)(o[nt][r] * inv);
    }
}

// ---------------------------------------------------------------------------
extern "C" void kernel_launch(void* const* d_in, const int* in_sizes, int n_in,
                              void* d_out, int out_size, void* d_ws, size_t ws_size,
                              hipStream_t stream)
{
    const float* x   = (const float*)d_in[0];
    const float* Wq  = (const float*)d_in[1];
    const float* bq  = (const float*)d_in[2];
    const float* Wk  = (const float*)d_in[3];
    const float* bk  = (const float*)d_in[4];
    const float* Wv  = (const float*)d_in[5];
    const float* bv  = (const float*)d_in[6];
    const float* Wo  = (const float*)d_in[7];
    const float* bo  = (const float*)d_in[8];
    const float* g1  = (const float*)d_in[9];
    const float* be1 = (const float*)d_in[10];
    const float* g2  = (const float*)d_in[11];
    const float* be2 = (const float*)d_in[12];
    const float* W1  = (const float*)d_in[13];
    const float* b1  = (const float*)d_in[14];
    const float* W2  = (const float*)d_in[15];
    const float* b2  = (const float*)d_in[16];

    char* ws = (char*)d_ws;
    constexpr size_t MB = 1024ull * 1024ull;
    // live-range based layout (aliases marked)
    h16*   wqT   = (h16*)(ws + 0 * MB);    // 2 MB
    h16*   wkT   = (h16*)(ws + 2 * MB);    // 2 MB
    h16*   wvT   = (h16*)(ws + 4 * MB);    // 2 MB
    h16*   woT   = (h16*)(ws + 6 * MB);    // 2 MB
    h16*   w1T   = (h16*)(ws + 8 * MB);    // 8 MB
    h16*   w2T   = (h16*)(ws + 16 * MB);   // 8 MB
    h16*   xnh   = (h16*)(ws + 24 * MB);   // 32 MB (reused for xn2h)
    float* qf    = (float*)(ws + 56 * MB); // 64 MB (reused for attnh)
    float* kf    = (float*)(ws + 120 * MB);// 64 MB (reused for res1)
    float* vf    = (float*)(ws + 184 * MB);// 64 MB (reused by ffn1h)
    h16*   qh    = (h16*)(ws + 248 * MB);  // 32 MB (reused by ffn1h)
    h16*   kh    = (h16*)(ws + 280 * MB);  // 32 MB (reused by ffn1h)
    h16*   vTh   = (h16*)(ws + 312 * MB);  // 32 MB -> total 344 MB
    h16*   attnh = (h16*)(ws + 56 * MB);   // alias qf (dead after rope)
    float* res1  = (float*)(ws + 120 * MB);// alias kf (dead after rope)
    h16*   ffn1h = (h16*)(ws + 184 * MB);  // alias vf+qh+kh (dead after attn), 128 MB
    h16*   xn2h  = (h16*)(ws + 24 * MB);   // alias xnh (dead after QKV GEMMs)

    const dim3 b256(256), b128(128), bT(32, 8);
    const dim3 gH (Hc  / 128, Mrows / 128);
    const dim3 gH4(H4c / 128, Mrows / 128);

    // 1) LN1 -> f16
    ln_cast_kernel<<<Mrows, b256, 0, stream>>>(x, g1, be1, xnh);
    // 2) weight transposes (f32 [K,N] -> f16 [N,K])
    wtrans_kernel<<<dim3(Hc / 32, Hc / 32), bT, 0, stream>>>(Wq, wqT, Hc, Hc);
    wtrans_kernel<<<dim3(Hc / 32, Hc / 32), bT, 0, stream>>>(Wk, wkT, Hc, Hc);
    wtrans_kernel<<<dim3(Hc / 32, Hc / 32), bT, 0, stream>>>(Wv, wvT, Hc, Hc);
    wtrans_kernel<<<dim3(Hc / 32, Hc / 32), bT, 0, stream>>>(Wo, woT, Hc, Hc);
    wtrans_kernel<<<dim3(H4c / 32, Hc / 32), bT, 0, stream>>>(W1, w1T, Hc, H4c);
    wtrans_kernel<<<dim3(Hc / 32, H4c / 32), bT, 0, stream>>>(W2, w2T, H4c, Hc);
    // 3) QKV projections (fp32 out, bias)
    gemm_f16_kernel<EPI_NONE, true, false><<<gH, b256, 0, stream>>>(
        xnh, wqT, bq, nullptr, qf, nullptr, Mrows, Hc, Hc);
    gemm_f16_kernel<EPI_NONE, true, false><<<gH, b256, 0, stream>>>(
        xnh, wkT, bk, nullptr, kf, nullptr, Mrows, Hc, Hc);
    gemm_f16_kernel<EPI_NONE, true, false><<<gH, b256, 0, stream>>>(
        xnh, wvT, bv, nullptr, vf, nullptr, Mrows, Hc, Hc);
    // 4) RoPE + head-major pack (v stored transposed)
    rope_pack_kernel<<<Mrows, b256, 0, stream>>>(qf, kf, vf, qh, kh, vTh);
    // 5) flash attention
    attn_kernel<<<dim3(Bc * NHc, Tc / 64), b128, 0, stream>>>(qh, kh, vTh, attnh);
    // 6) output projection + residual -> res1 (fp32)
    gemm_f16_kernel<EPI_RESID, true, false><<<gH, b256, 0, stream>>>(
        attnh, woT, bo, x, res1, nullptr, Mrows, Hc, Hc);
    // 7) LN2 -> f16
    ln_cast_kernel<<<Mrows, b256, 0, stream>>>(res1, g2, be2, xn2h);
    // 8) FFN1 + exact GELU -> f16
    gemm_f16_kernel<EPI_GELU, false, true><<<gH4, b256, 0, stream>>>(
        xn2h, w1T, b1, nullptr, nullptr, ffn1h, Mrows, H4c, Hc);
    // 9) FFN2 + residual -> fp32 output
    gemm_f16_kernel<EPI_RESID, true, false><<<gH, b256, 0, stream>>>(
        ffn1h, w2T, b2, res1, (float*)d_out, nullptr, Mrows, Hc, H4c);
}
